// MKEnsemble_88510686036868
// MI455X (gfx1250) — compile-verified, matrix operand order checked by reading, not run on hardware
//
#include <hip/hip_runtime.h>
#include <math.h>

#define NNODES 524288
#define NEDGES 2097152
#define NFRAGS 65536
#define NMOLS  8192
#define DD     128
#define HH     256
#define NEXP   8
#define NTASK  3

typedef __bf16 bf16_t;
typedef bf16_t v16bf __attribute__((ext_vector_type(16)));
typedef float  v8f   __attribute__((ext_vector_type(8)));

union WFrag { v16bf v; unsigned u[8]; };

// Load a 16x32 bf16 WMMA operand tile from row-major memory (row stride ld elems).
// Lane layout per CDNA5 ISA 7.12.2: lanes 0-15 = rows 0-15 with K 0..7,16..23;
// lanes 16-31 = rows 0-15 with K 8..15,24..31. Each VGPR packs 2 consecutive K.
__device__ __forceinline__ void load_frag(WFrag& f, const bf16_t* __restrict__ base,
                                          int ld, int lane) {
  int r  = lane & 15;
  int kb = (lane >> 4) * 8;
  const bf16_t* p = base + r * ld + kb;
#pragma unroll
  for (int i = 0; i < 8; ++i) {
    int k = 2 * i + ((i >= 4) ? 8 : 0);
    f.u[i] = *reinterpret_cast<const unsigned*>(p + k);
  }
}

__device__ __forceinline__ float gelu_exact(float x) {
  return 0.5f * x * (1.0f + erff(x * 0.70710678118654752f));
}

// ---------------- weight convert + transpose: src [B][K][N] f32 -> dst [B][N][K] bf16
__global__ void __launch_bounds__(256) convt_kernel(const float* __restrict__ src,
                                                    bf16_t* __restrict__ dst,
                                                    int K, int N, int total) {
  int idx = blockIdx.x * 256 + threadIdx.x;
  if (idx >= total) return;
  int kn = K * N;
  int b = idx / kn;
  int rem = idx - b * kn;
  int k = rem / N;
  int n = rem - k * N;
  dst[b * kn + n * K + k] = (bf16_t)src[idx];
}

// ---------------- edge scatter-sum: agg[dst] += h[src]
__global__ void __launch_bounds__(256) edge_agg_kernel(const float* __restrict__ h,
                                                       const int* __restrict__ ei,
                                                       float* __restrict__ agg) {
  int id = blockIdx.x * 256 + threadIdx.x;      // NEDGES*32 threads
  int e = id >> 5;
  int lane = id & 31;
  int s = ei[e];
  int d = ei[NEDGES + e];
  const float* sp = h + s * DD + lane * 4;
  float* dp = agg + d * DD + lane * 4;
  atomicAdd(dp + 0, sp[0]);
  atomicAdd(dp + 1, sp[1]);
  atomicAdd(dp + 2, sp[2]);
  atomicAdd(dp + 3, sp[3]);
}

// ---------------- fused GIN MLP: z=(1+eps)h+agg; relu(z@W1+b1)@W2+b2; BN; relu
__global__ void __launch_bounds__(256) gin_mlp_kernel(
    const float* __restrict__ hin, const float* __restrict__ agg,
    const float* __restrict__ eps3, int layer,
    const bf16_t* __restrict__ w1t, const bf16_t* __restrict__ w2t,
    const float* __restrict__ b1, const float* __restrict__ b2,
    const float* __restrict__ gamma, const float* __restrict__ beta,
    float* __restrict__ hout) {
  __shared__ bf16_t zt[16 * DD];
  __shared__ bf16_t mid[16 * DD];
  const int tid = threadIdx.x;
  const int lane = tid & 31;
  const int wave = tid >> 5;
  const int m0 = blockIdx.x * 16;
  const float eps1 = 1.0f + eps3[layer];
#pragma unroll
  for (int i = 0; i < 8; ++i) {            // stage z tile as bf16 into LDS
    int idx = tid + i * 256;
    int g = m0 * DD + idx;
    zt[idx] = (bf16_t)(eps1 * hin[g] + agg[g]);
  }
  __syncthreads();
  const int n0 = wave * 16;
  v8f acc = {};
#pragma unroll
  for (int k0 = 0; k0 < DD; k0 += 32) {
    WFrag a, b;
    load_frag(a, zt + k0, DD, lane);
    load_frag(b, w1t + n0 * DD + k0, DD, lane);
    acc = __builtin_amdgcn_wmma_f32_16x16x32_bf16(false, a.v, false, b.v,
                                                  (short)0, acc, false, false);
  }
  const int cn = lane & 15;
  const int mh = (lane >> 4) * 8;
#pragma unroll
  for (int v = 0; v < 8; ++v) {
    float y = acc[v] + b1[n0 + cn];
    mid[(mh + v) * DD + n0 + cn] = (bf16_t)fmaxf(y, 0.0f);
  }
  __syncthreads();
  v8f acc2 = {};
#pragma unroll
  for (int k0 = 0; k0 < DD; k0 += 32) {
    WFrag a, b;
    load_frag(a, mid + k0, DD, lane);
    load_frag(b, w2t + n0 * DD + k0, DD, lane);
    acc2 = __builtin_amdgcn_wmma_f32_16x16x32_bf16(false, a.v, false, b.v,
                                                   (short)0, acc2, false, false);
  }
  const float inv_std = 0.9999950000374997f;  // 1/sqrt(1+1e-5)
#pragma unroll
  for (int v = 0; v < 8; ++v) {
    int col = n0 + cn;
    float y = acc2[v] + b2[col];
    y = y * inv_std * gamma[col] + beta[col];
    hout[(m0 + mh + v) * DD + col] = fmaxf(y, 0.0f);
  }
}

// ---------------- segment sum pooling via atomics (32 lanes x 4 floats per row)
__global__ void __launch_bounds__(256) seg_pool_kernel(const float* __restrict__ vals,
                                                       const int* __restrict__ seg,
                                                       float* __restrict__ sum,
                                                       float* __restrict__ cnt) {
  int id = blockIdx.x * 256 + threadIdx.x;    // rows*32 threads
  int row = id >> 5;
  int lane = id & 31;
  int s = seg[row];
  const float* sp = vals + row * DD + lane * 4;
  float* dp = sum + s * DD + lane * 4;
  atomicAdd(dp + 0, sp[0]);
  atomicAdd(dp + 1, sp[1]);
  atomicAdd(dp + 2, sp[2]);
  atomicAdd(dp + 3, sp[3]);
  if (lane == 0) atomicAdd(&cnt[s], 1.0f);
}

__global__ void __launch_bounds__(256) seg_div_kernel(const float* __restrict__ sum,
                                                      const float* __restrict__ cnt,
                                                      float* __restrict__ outf,
                                                      bf16_t* __restrict__ outb,
                                                      int total) {
  int idx = blockIdx.x * 256 + threadIdx.x;
  if (idx >= total) return;
  float c = fmaxf(cnt[idx >> 7], 1.0f);
  float v = sum[idx] / c;
  outf[idx] = v;
  if (outb) outb[idx] = (bf16_t)v;
}

// ---------------- router: top-2 softmax weights + load accumulation
__global__ void __launch_bounds__(256) router_kernel(const float* __restrict__ femb,
                                                     const float* __restrict__ gate,
                                                     float* __restrict__ wts,
                                                     float* __restrict__ loadsum) {
  __shared__ float gs[DD * NEXP];
  __shared__ float ls[NEXP];
  int tid = threadIdx.x;
  for (int i = tid; i < DD * NEXP; i += 256) gs[i] = gate[i];
  if (tid < NEXP) ls[tid] = 0.0f;
  __syncthreads();
  int f = blockIdx.x * 256 + tid;
  float lg[NEXP];
#pragma unroll
  for (int e = 0; e < NEXP; ++e) lg[e] = 0.0f;
  const float* xr = femb + f * DD;
  for (int d = 0; d < DD; ++d) {
    float xv = xr[d];
#pragma unroll
    for (int e = 0; e < NEXP; ++e) lg[e] = fmaf(xv, gs[d * NEXP + e], lg[e]);
  }
  int i1 = 0;
#pragma unroll
  for (int e = 1; e < NEXP; ++e) if (lg[e] > lg[i1]) i1 = e;
  int i2 = (i1 == 0) ? 1 : 0;
#pragma unroll
  for (int e = 0; e < NEXP; ++e) if (e != i1 && lg[e] > lg[i2]) i2 = e;
  float e2 = expf(lg[i2] - lg[i1]);
  float den = 1.0f + e2;
  float w1 = 1.0f / den, w2 = e2 / den;
  float* wr = wts + f * NEXP;
#pragma unroll
  for (int e = 0; e < NEXP; ++e) wr[e] = 0.0f;
  wr[i1] = w1;
  wr[i2] = w2;
  atomicAdd(&ls[i1], w1);
  atomicAdd(&ls[i2], w2);
  __syncthreads();
  if (tid < NEXP) atomicAdd(&loadsum[tid], ls[tid]);
}

// ---------------- register-blocked expert GEMM: out = act(A @ Wt^T + bias)
// Block: 16 rows (A tile staged once in LDS via async global->LDS DMA),
// 8 waves x NTPW 16-col tiles = N cols.
// MODE 0: GELU -> bf16 out [M,N]; MODE 1: frag_moe[f] += wts[f][expert]*val (N==128)
template <int NTPW, int MODE>
__global__ void __launch_bounds__(256) gemm_wmma_kernel(
    const bf16_t* __restrict__ A, const bf16_t* __restrict__ Wt,
    const float* __restrict__ bias, int K,
    bf16_t* __restrict__ outB, float* __restrict__ outAcc,
    const float* __restrict__ gwts, int expert) {
  constexpr int N = NTPW * 128;
  __shared__ bf16_t As[16 * HH];           // up to K=256: 8KB
  const int tid = threadIdx.x;
  const int lane = tid & 31;
  const int wave = tid >> 5;
  const int m0 = blockIdx.x * 16;
  // Async stage of the 16xK A tile: GLOBAL_LOAD_ASYNC_TO_LDS_B128, 16B per lane,
  // tracked by ASYNCcnt (CDNA5 ISA 10.7 / 15.18.3). (16*K*2)/16 is a multiple of
  // 256 for K in {128,256}, so every lane is active on every issue.
  {
    const unsigned long long gbase =
        (unsigned long long)(size_t)(A + (size_t)m0 * K);
    // generic->LDS offset: LDS aperture keeps the offset in addr[31:0]
    const unsigned lds_base = (unsigned)(size_t)(&As[0]);
    const int nvec = (16 * K) >> 3;        // 16-byte chunks
    for (int i = tid; i < nvec; i += 256) {
      unsigned long long ga = gbase + (unsigned long long)i * 16u;
      unsigned la = lds_base + (unsigned)i * 16u;
      asm volatile("global_load_async_to_lds_b128 %0, %1, off"
                   :
                   : "v"(la), "v"(ga)
                   : "memory");
    }
    asm volatile("s_wait_asynccnt 0x0" ::: "memory");
  }
  __syncthreads();
  v8f acc[NTPW];
#pragma unroll
  for (int t = 0; t < NTPW; ++t)
#pragma unroll
    for (int j = 0; j < 8; ++j) acc[t][j] = 0.0f;
  const int nbase = wave * 16 * NTPW;
  for (int k0 = 0; k0 < K; k0 += 32) {
    WFrag a;
    load_frag(a, As + k0, K, lane);
#pragma unroll
    for (int t = 0; t < NTPW; ++t) {
      WFrag b;
      load_frag(b, Wt + (size_t)(nbase + t * 16) * K + k0, K, lane);
      acc[t] = __builtin_amdgcn_wmma_f32_16x16x32_bf16(false, a.v, false, b.v,
                                                       (short)0, acc[t], false, false);
    }
  }
  const int cn = lane & 15;
  const int mh = (lane >> 4) * 8;
#pragma unroll
  for (int t = 0; t < NTPW; ++t) {
#pragma unroll
    for (int v = 0; v < 8; ++v) {
      int row = m0 + mh + v;
      int col = nbase + t * 16 + cn;
      float y = acc[t][v] + bias[col];
      if (MODE == 0) {
        outB[(size_t)row * N + col] = (bf16_t)gelu_exact(y);
      } else {
        outAcc[(size_t)row * DD + col] += gwts[row * NEXP + expert] * y;
      }
    }
  }
}

// ---------------- task heads: Linear(128,64)+ReLU+Linear(64,1)
__global__ void __launch_bounds__(256) heads_kernel(const float* __restrict__ memb,
                                                    const float* __restrict__ hw1,
                                                    const float* __restrict__ hb1,
                                                    const float* __restrict__ hw2,
                                                    const float* __restrict__ hb2,
                                                    float* __restrict__ out) {
  __shared__ float w1[DD * 64];
  __shared__ float b1s[64], w2s[64];
  int t = blockIdx.y;
  int tid = threadIdx.x;
  for (int i = tid; i < DD * 64; i += 256) w1[i] = hw1[t * DD * 64 + i];
  if (tid < 64) { b1s[tid] = hb1[t * 64 + tid]; w2s[tid] = hw2[t * 64 + tid]; }
  __syncthreads();
  int m = blockIdx.x * 256 + tid;
  const float* xr = memb + m * DD;
  float accv = 0.0f;
  for (int j = 0; j < 64; ++j) {
    float s = b1s[j];
#pragma unroll 8
    for (int d = 0; d < DD; ++d) s = fmaf(xr[d], w1[d * 64 + j], s);
    accv = fmaf(fmaxf(s, 0.0f), w2s[j], accv);
  }
  out[m * NTASK + t] = accv + hb2[t];
}

__global__ void lb_kernel(const float* __restrict__ loadsum, float* __restrict__ out) {
  if (threadIdx.x == 0 && blockIdx.x == 0) {
    float s = 0.0f;
    for (int e = 0; e < NEXP; ++e) {
      float l = loadsum[e] / (float)NFRAGS;
      s += l * l;
    }
    out[NMOLS * NTASK] = (float)NEXP * s;
  }
}

extern "C" void kernel_launch(void* const* d_in, const int* in_sizes, int n_in,
                              void* d_out, int out_size, void* d_ws, size_t ws_size,
                              hipStream_t stream) {
  const float* x        = (const float*)d_in[0];
  const float* gin_w1   = (const float*)d_in[1];
  const float* gin_b1   = (const float*)d_in[2];
  const float* gin_w2   = (const float*)d_in[3];
  const float* gin_b2   = (const float*)d_in[4];
  const float* gin_eps  = (const float*)d_in[5];
  const float* bn_gamma = (const float*)d_in[6];
  const float* bn_beta  = (const float*)d_in[7];
  const float* gate_w   = (const float*)d_in[8];
  const float* ew1      = (const float*)d_in[9];
  const float* eb1      = (const float*)d_in[10];
  const float* ew2      = (const float*)d_in[11];
  const float* eb2      = (const float*)d_in[12];
  const float* ew3      = (const float*)d_in[13];
  const float* eb3      = (const float*)d_in[14];
  const float* hw1      = (const float*)d_in[15];
  const float* hb1      = (const float*)d_in[16];
  const float* hw2      = (const float*)d_in[17];
  const float* hb2      = (const float*)d_in[18];
  const int* edge_index = (const int*)d_in[19];
  const int* node2frag  = (const int*)d_in[20];
  const int* mol_idx    = (const int*)d_in[21];

  // ---- workspace carve (256B aligned bump allocator)
  char* base = (char*)d_ws;
  size_t off = 0;
  auto alloc = [&](size_t bytes) -> void* {
    off = (off + 255) & ~(size_t)255;
    void* p = base + off;
    off += bytes;
    return p;
  };
  bf16_t* gw1t   = (bf16_t*)alloc((size_t)3 * DD * DD * 2);
  bf16_t* gw2t   = (bf16_t*)alloc((size_t)3 * DD * DD * 2);
  bf16_t* e1t    = (bf16_t*)alloc((size_t)NEXP * DD * HH * 2);
  bf16_t* e2t    = (bf16_t*)alloc((size_t)NEXP * HH * HH * 2);
  bf16_t* e3t    = (bf16_t*)alloc((size_t)NEXP * HH * DD * 2);
  float*  hA     = (float*)alloc((size_t)NNODES * DD * 4);
  float*  hB     = (float*)alloc((size_t)NNODES * DD * 4);
  float*  agg    = (float*)alloc((size_t)NNODES * DD * 4);
  float*  fsum   = (float*)alloc((size_t)NFRAGS * DD * 4);
  float*  fcnt   = (float*)alloc((size_t)NFRAGS * 4);
  float*  femb   = (float*)alloc((size_t)NFRAGS * DD * 4);
  bf16_t* fembb  = (bf16_t*)alloc((size_t)NFRAGS * DD * 2);
  float*  wts    = (float*)alloc((size_t)NFRAGS * NEXP * 4);
  float*  loadsum= (float*)alloc(NEXP * 4);
  bf16_t* a1     = (bf16_t*)alloc((size_t)NFRAGS * HH * 2);
  bf16_t* a2     = (bf16_t*)alloc((size_t)NFRAGS * HH * 2);
  float*  fmoe   = (float*)alloc((size_t)NFRAGS * DD * 4);
  float*  msum   = (float*)alloc((size_t)NMOLS * DD * 4);
  float*  mcnt   = (float*)alloc((size_t)NMOLS * 4);
  float*  memb   = (float*)alloc((size_t)NMOLS * DD * 4);
  (void)ws_size; (void)in_sizes; (void)n_in; (void)out_size;

  // ---- convert + transpose weights to bf16 [N][K]
  {
    int t1 = 3 * DD * DD;
    convt_kernel<<<(t1 + 255) / 256, 256, 0, stream>>>(gin_w1, gw1t, DD, DD, t1);
    convt_kernel<<<(t1 + 255) / 256, 256, 0, stream>>>(gin_w2, gw2t, DD, DD, t1);
    int t2 = NEXP * DD * HH;
    convt_kernel<<<(t2 + 255) / 256, 256, 0, stream>>>(ew1, e1t, DD, HH, t2);
    int t3 = NEXP * HH * HH;
    convt_kernel<<<(t3 + 255) / 256, 256, 0, stream>>>(ew2, e2t, HH, HH, t3);
    int t4 = NEXP * HH * DD;
    convt_kernel<<<(t4 + 255) / 256, 256, 0, stream>>>(ew3, e3t, HH, DD, t4);
  }

  // ---- GIN encoder (3 layers)
  const float* hin = x;
  float* ping[2] = {hA, hB};
  for (int l = 0; l < 3; ++l) {
    hipMemsetAsync(agg, 0, (size_t)NNODES * DD * 4, stream);
    edge_agg_kernel<<<(NEDGES * 32) / 256, 256, 0, stream>>>(hin, edge_index, agg);
    float* ho = ping[l & 1];
    gin_mlp_kernel<<<NNODES / 16, 256, 0, stream>>>(
        hin, agg, gin_eps, l, gw1t + l * DD * DD, gw2t + l * DD * DD,
        gin_b1 + l * DD, gin_b2 + l * DD, bn_gamma + l * DD, bn_beta + l * DD, ho);
    hin = ho;
  }

  // ---- fragment mean pool
  hipMemsetAsync(fsum, 0, (size_t)NFRAGS * DD * 4, stream);
  hipMemsetAsync(fcnt, 0, (size_t)NFRAGS * 4, stream);
  seg_pool_kernel<<<(NNODES * 32) / 256, 256, 0, stream>>>(hin, node2frag, fsum, fcnt);
  seg_div_kernel<<<(NFRAGS * DD + 255) / 256, 256, 0, stream>>>(fsum, fcnt, femb, fembb,
                                                                NFRAGS * DD);

  // ---- router
  hipMemsetAsync(loadsum, 0, NEXP * 4, stream);
  router_kernel<<<NFRAGS / 256, 256, 0, stream>>>(femb, gate_w, wts, loadsum);

  // ---- experts (dense, all 8), weighted accumulate into frag_moe
  hipMemsetAsync(fmoe, 0, (size_t)NFRAGS * DD * 4, stream);
  for (int e = 0; e < NEXP; ++e) {
    gemm_wmma_kernel<2, 0><<<NFRAGS / 16, 256, 0, stream>>>(
        fembb, e1t + (size_t)e * DD * HH, eb1 + e * HH, DD, a1, nullptr,
        nullptr, e);
    gemm_wmma_kernel<2, 0><<<NFRAGS / 16, 256, 0, stream>>>(
        a1, e2t + (size_t)e * HH * HH, eb2 + e * HH, HH, a2, nullptr,
        nullptr, e);
    gemm_wmma_kernel<1, 1><<<NFRAGS / 16, 256, 0, stream>>>(
        a2, e3t + (size_t)e * HH * DD, eb3 + e * DD, HH, nullptr, fmoe,
        wts, e);
  }

  // ---- molecule mean pool
  hipMemsetAsync(msum, 0, (size_t)NMOLS * DD * 4, stream);
  hipMemsetAsync(mcnt, 0, (size_t)NMOLS * 4, stream);
  seg_pool_kernel<<<(NFRAGS * 32) / 256, 256, 0, stream>>>(fmoe, mol_idx, msum, mcnt);
  seg_div_kernel<<<(NMOLS * DD + 255) / 256, 256, 0, stream>>>(msum, mcnt, memb,
                                                               nullptr, NMOLS * DD);

  // ---- heads + lb loss
  heads_kernel<<<dim3(NMOLS / 256, NTASK), 256, 0, stream>>>(memb, hw1, hb1, hw2, hb2,
                                                             (float*)d_out);
  lb_kernel<<<1, 64, 0, stream>>>(loadsum, (float*)d_out);
}
